// CRLoss_52664888983707
// MI455X (gfx1250) — matrix operand order repaired and checked by reading, not compiled, not run on hardware
//
#include <hip/hip_runtime.h>

typedef __attribute__((ext_vector_type(16))) _Float16 v16h;
typedef __attribute__((ext_vector_type(8)))  _Float16 v8h;
typedef __attribute__((ext_vector_type(8)))  float    v8f;
typedef __attribute__((ext_vector_type(4)))  float    v4f;
typedef __attribute__((ext_vector_type(4)))  int      v4i;

#define BDIM 8192
#define DDIM 512

// ---------------------------------------------------------------------------
// Kernel 1: row-normalize img/txt/txt_cr to f16, compute sm, sm_cr, margin_cr.
// One wave (32 lanes) per row; 16 f32 elements per lane.
// ---------------------------------------------------------------------------
__global__ __launch_bounds__(256)
void crloss_normalize(const float* __restrict__ img, const float* __restrict__ txt,
                      const float* __restrict__ txc, const float* __restrict__ marg,
                      _Float16* __restrict__ an, _Float16* __restrict__ bn,
                      _Float16* __restrict__ cn,
                      float* __restrict__ smv, float* __restrict__ smcv,
                      float* __restrict__ mgcv)
{
    const int wave = threadIdx.x >> 5;
    const int lane = threadIdx.x & 31;
    const int row  = blockIdx.x * 8 + wave;
    const size_t base = (size_t)row * DDIM + lane * 16;

    float av[16], bv[16], cv[16];
    float aa = 0.f, bb = 0.f, cc = 0.f, ab = 0.f, ac = 0.f;
#pragma unroll
    for (int q = 0; q < 4; ++q) {
        v4f a = *(const v4f*)(img + base + q * 4);
        v4f b = *(const v4f*)(txt + base + q * 4);
        v4f c = *(const v4f*)(txc + base + q * 4);
#pragma unroll
        for (int t = 0; t < 4; ++t) {
            av[q * 4 + t] = a[t]; bv[q * 4 + t] = b[t]; cv[q * 4 + t] = c[t];
            aa += a[t] * a[t]; bb += b[t] * b[t]; cc += c[t] * c[t];
            ab += a[t] * b[t]; ac += a[t] * c[t];
        }
    }
#pragma unroll
    for (int off = 16; off > 0; off >>= 1) {
        aa += __shfl_xor(aa, off, 32);
        bb += __shfl_xor(bb, off, 32);
        cc += __shfl_xor(cc, off, 32);
        ab += __shfl_xor(ab, off, 32);
        ac += __shfl_xor(ac, off, 32);
    }
    const float ia = 1.f / (sqrtf(aa) + 1e-8f);
    const float ib = 1.f / (sqrtf(bb) + 1e-8f);
    const float ic = 1.f / (sqrtf(cc) + 1e-8f);

    v8h ha0, ha1, hb0, hb1, hc0, hc1;
#pragma unroll
    for (int t = 0; t < 8; ++t) {
        ha0[t] = (_Float16)(av[t] * ia);      ha1[t] = (_Float16)(av[t + 8] * ia);
        hb0[t] = (_Float16)(bv[t] * ib);      hb1[t] = (_Float16)(bv[t + 8] * ib);
        hc0[t] = (_Float16)(cv[t] * ic);      hc1[t] = (_Float16)(cv[t + 8] * ic);
    }
    *(v8h*)(an + base)     = ha0;  *(v8h*)(an + base + 8) = ha1;
    *(v8h*)(bn + base)     = hb0;  *(v8h*)(bn + base + 8) = hb1;
    *(v8h*)(cn + base)     = hc0;  *(v8h*)(cn + base + 8) = hc1;

    if (lane == 0) {
        const float sm  = ab * (ia * ib);
        const float smc = ac * (ia * ic);
        smv[row]  = sm;
        smcv[row] = smc;
        float lam = fabsf(smc) / fabsf(sm);
        lam = fminf(lam, 1.0f);
        mgcv[row] = (lam + 1.0f) * marg[row] * 0.5f;
    }
}

// ---------------------------------------------------------------------------
// Kernel 2: fused GEMM (A @ B^T via v_wmma_f32_16x16x32_f16) + semi-hard
// negative mining. One wave owns 16 rows, loops over all 8192 columns.
// - A fragments register-resident (16 x v16h).
// - Labels staged in LDS so the mining read is DScnt-tracked and never
//   drains the pipelined LOADcnt B-fragment loads.
// - Two independent WMMA accumulation chains (even/odd K-chunks).
// First-valid-negative = min column index; tracked as packed (j<<32|bits(s))
// per lane, reduced across the 16-lane group once at the end.
// ---------------------------------------------------------------------------
__global__ __launch_bounds__(256)
void crloss_mine(const _Float16* __restrict__ Am, const _Float16* __restrict__ Bm,
                 const float* __restrict__ diagv, const float* __restrict__ mgv,
                 const int* __restrict__ labels, const int* __restrict__ aflag,
                 float* __restrict__ partial)
{
    __shared__ int slab[BDIM];                     // 32 KB of the 320 KB WGP LDS

    const int wave  = threadIdx.x >> 5;
    const int lane  = threadIdx.x & 31;
    const int half  = lane >> 4;        // 0: K 0-7/16-23 (rows 0-7), 1: K 8-15/24-31 (rows 8-15)
    const int l16   = lane & 15;
    const int row0  = blockIdx.x * 128 + wave * 16;
    const int rbase = half * 8;

    // stage labels into LDS (cooperative, vectorized)
    for (int t = threadIdx.x * 4; t < BDIM; t += 256 * 4)
        *(v4i*)(slab + t) = *(const v4i*)(labels + t);

    // per-lane metadata for the 8 rows this half-wave owns in C/D layout
    float smr[8], mgr[8]; int lbr[8];
#pragma unroll
    for (int r = 0; r < 8; ++r) {
        const int i = row0 + rbase + r;
        smr[r] = diagv[i];
        mgr[r] = mgv[i];
        lbr[r] = labels[i];
    }

    // Resident A fragments: 16 K-chunks of the 16x512 row slab.
    // ISA 16-bit A layout: lanes 0-15 -> K {0..7, 16..23}, lanes 16-31 -> K {8..15, 24..31}.
    v16h afrag[16];
    {
        const _Float16* arow = Am + (size_t)(row0 + l16) * DDIM;
        const int ka = half * 8;
#pragma unroll
        for (int c = 0; c < 16; ++c) {
            union { v16h v; v8h h[2]; } u;
            u.h[0] = *(const v8h*)(arow + c * 32 + ka);
            u.h[1] = *(const v8h*)(arow + c * 32 + ka + 16);
            afrag[c] = u.v;
        }
    }

    __syncthreads();                               // labels visible to all waves

    unsigned long long best[8];
#pragma unroll
    for (int r = 0; r < 8; ++r) best[r] = ~0ull;

    const int kb = half * 8;    // B layout: lanes 0-15 -> K {0..7, 16..23}, lanes 16-31 -> K {8..15, 24..31}

    for (int j0 = 0; j0 < BDIM; j0 += 16) {
        const _Float16* brow = Bm + (size_t)(j0 + l16) * DDIM;
        if (j0 + 16 < BDIM)
            __builtin_prefetch(brow + (size_t)16 * DDIM, 0, 0);

        v8f acc0 = {}, acc1 = {};                  // two independent WMMA chains
#pragma unroll
        for (int c = 0; c < 16; c += 2) {
            union { v16h v; v8h h[2]; } u0, u1;
            u0.h[0] = *(const v8h*)(brow + (c + 0) * 32 + kb);
            u0.h[1] = *(const v8h*)(brow + (c + 0) * 32 + kb + 16);
            u1.h[0] = *(const v8h*)(brow + (c + 1) * 32 + kb);
            u1.h[1] = *(const v8h*)(brow + (c + 1) * 32 + kb + 16);
            acc0 = __builtin_amdgcn_wmma_f32_16x16x32_f16(
                false, afrag[c + 0], false, u0.v, (short)0, acc0, false, false);
            acc1 = __builtin_amdgcn_wmma_f32_16x16x32_f16(
                false, afrag[c + 1], false, u1.v, (short)0, acc1, false, false);
        }
        const v8f acc = acc0 + acc1;

        const int jc = j0 + l16;                   // this lane's column in C/D layout
        const int lj = slab[jc];                   // DScnt-tracked, no LOADcnt drain
#pragma unroll
        for (int r = 0; r < 8; ++r) {
            const float s  = acc[r];
            const float lm = s - smr[r] + mgr[r];
            const bool valid = (lj != lbr[r]) && (lm > 0.f) && (lm < mgr[r]);
            const unsigned long long key =
                ((unsigned long long)(unsigned)jc << 32) |
                (unsigned long long)__float_as_uint(s);
            if (valid && key < best[r]) best[r] = key;
        }
    }

    // cross-lane min (first valid negative) within each 16-lane group
#pragma unroll
    for (int r = 0; r < 8; ++r) {
        unsigned long long k = best[r];
#pragma unroll
        for (int off = 1; off < 16; off <<= 1) {
            unsigned long long o = __shfl_xor(k, off, 32);
            if (o < k) k = o;
        }
        best[r] = k;
    }

    const int aflg = *aflag;
    float loss = 0.f;
#pragma unroll
    for (int r = 0; r < 8; ++r) {
        if (best[r] != ~0ull) {
            const bool ok = aflg ? (mgr[r] >= 0.16f) : true;
            if (ok) {
                const float s = __uint_as_float((unsigned)(best[r] & 0xffffffffu));
                const float v = mgr[r] - smr[r] + s;
                loss += (v > 0.f) ? v : 0.f;
            }
        }
    }
    // lane 0 holds rows 0-7 sum, lane 16 holds rows 8-15 sum (uniform per group)
    const float other = __shfl_xor(loss, 16, 32);
    if (lane == 0) partial[blockIdx.x * 8 + wave] = loss + other;
}

// ---------------------------------------------------------------------------
// Kernel 3: deterministic fixed-order reduction of 4*512 partials,
// out = (base0+base1) + cr_beta*(cr0+cr1)
// ---------------------------------------------------------------------------
__global__ __launch_bounds__(256)
void crloss_finalize(const float* __restrict__ partial, const float* __restrict__ crbeta,
                     float* __restrict__ out)
{
    __shared__ float sb[256], sc[256];
    float b = 0.f, c = 0.f;
    for (int t = threadIdx.x; t < 1024; t += 256) b += partial[t];
    for (int t = threadIdx.x; t < 1024; t += 256) c += partial[1024 + t];
    sb[threadIdx.x] = b; sc[threadIdx.x] = c;
    __syncthreads();
    for (int s = 128; s > 0; s >>= 1) {
        if (threadIdx.x < (unsigned)s) {
            sb[threadIdx.x] += sb[threadIdx.x + s];
            sc[threadIdx.x] += sc[threadIdx.x + s];
        }
        __syncthreads();
    }
    if (threadIdx.x == 0) out[0] = sb[0] + crbeta[0] * sc[0];
}

// ---------------------------------------------------------------------------
extern "C" void kernel_launch(void* const* d_in, const int* in_sizes, int n_in,
                              void* d_out, int out_size, void* d_ws, size_t ws_size,
                              hipStream_t stream)
{
    const float* img = (const float*)d_in[0];
    const float* txt = (const float*)d_in[1];
    const float* txc = (const float*)d_in[2];
    const int*   lab = (const int*)  d_in[3];
    const int*   afl = (const int*)  d_in[4];
    const float* mrg = (const float*)d_in[5];
    const float* crb = (const float*)d_in[6];

    char* ws = (char*)d_ws;
    _Float16* an = (_Float16*)(ws);
    _Float16* bn = (_Float16*)(ws + ((size_t)8 << 20));
    _Float16* cn = (_Float16*)(ws + ((size_t)16 << 20));
    float* smv   = (float*)(ws + ((size_t)24 << 20));
    float* smc   = smv + BDIM;
    float* mgc   = smc + BDIM;
    float* part  = mgc + BDIM;           // 4 * 512 partial sums

    crloss_normalize<<<dim3(BDIM / 8), dim3(256), 0, stream>>>(
        img, txt, txc, mrg, an, bn, cn, smv, smc, mgc);

    const dim3 g(BDIM / 128), blk(256);
    // base: sim row-direction, then sim^T (swap operands), diag = sm, margin = input margin
    crloss_mine<<<g, blk, 0, stream>>>(an, bn, smv, mrg, lab, afl, part + 0);
    crloss_mine<<<g, blk, 0, stream>>>(bn, an, smv, mrg, lab, afl, part + 512);
    // cr: sim_cr and sim_cr^T, diag = sm_cr, margin = margin_cr
    crloss_mine<<<g, blk, 0, stream>>>(an, cn, smc, mgc, lab, afl, part + 1024);
    crloss_mine<<<g, blk, 0, stream>>>(cn, an, smc, mgc, lab, afl, part + 1536);

    crloss_finalize<<<dim3(1), dim3(256), 0, stream>>>(part, crb, (float*)d_out);
}
